// SinkhornCausalAttention_48533130445202
// MI455X (gfx1250) — matrix operand order repaired and checked by reading, not compiled, not run on hardware
//
#include <hip/hip_runtime.h>
#include <hip/hip_bf16.h>

#define T_LEN 4096
#define DH    128
#define NH    8
#define HH    4
#define NB    32
#define BS    128
#define ROT   (BS - 1)          // rotate amount for second-half heads
#define SCALE 0.03125f          // DIM^-0.5 = 1024^-0.5

typedef __attribute__((ext_vector_type(16))) __bf16 v16bf;
typedef __attribute__((ext_vector_type(8)))  float  v8f;

// LDS carve (dynamic shared): Q A-frags 32K | K' B-frags 64K (P A-frags overlay) |
// V' B-frags 64K | S f32 128K | R row 128B
#define SM_Q 0
#define SM_K 32768
#define SM_V 98304
#define SM_S 163840
#define SM_R 294912
#define SM_TOTAL 295040

__device__ __forceinline__ unsigned int f2bf1(float f) {
  unsigned int x = __float_as_uint(f);
  return (x + 0x7fffu + ((x >> 16) & 1u)) >> 16;   // RNE f32->bf16
}
__device__ __forceinline__ unsigned int f2bf2(float lo, float hi) {
  return f2bf1(lo) | (f2bf1(hi) << 16);
}
__device__ __forceinline__ v16bf ld_frag(const char* p) {
  union { uint4 u[2]; v16bf f; } t;
  t.u[0] = *(const uint4*)p;
  t.u[1] = *(const uint4*)(p + 16);
  return t.f;
}

// ---------------------------------------------------------------------------
// Kernel 1: bucket features -> Gumbel-softmax routing matrix R [32,32,32] f32
// ---------------------------------------------------------------------------
__global__ __launch_bounds__(128) void sort_r_kernel(
    const float* __restrict__ k, const float* __restrict__ sortw,
    const float* __restrict__ gum, float* __restrict__ R) {
  __shared__ float feat[NB * 256];
  __shared__ float lg[NB * NB];
  const int bh  = blockIdx.x;
  const int h   = bh & (NH - 1);
  const int rot = (h >= HH) ? ROT : 0;
  const int d   = threadIdx.x;              // 0..127, one feature dim
  const size_t base = (size_t)bh * T_LEN;

  float sum = 0.f;
  for (int t = 0; t < T_LEN; ++t) {
    float kv = k[(base + (size_t)((t + rot) & (T_LEN - 1))) * DH + d];
    sum += kv;
    if ((t & (BS - 1)) == 0) {              // first token of bucket u
      int u = t >> 7;
      feat[u * 256 + d]      = sum / (float)(t + 1);  // cumulative mean
      feat[u * 256 + DH + d] = kv;                    // raw key
    }
  }
  __syncthreads();

  const float* wp = sortw + (size_t)h * 256 * NB;
  for (int o = 0; o < 8; ++o) {
    int idx = d * 8 + o;                    // 0..1023
    int u = idx >> 5, j = idx & 31;
    float acc = 0.f;
    for (int e = 0; e < 256; ++e) acc += feat[u * 256 + e] * wp[e * NB + j];
    float rl = __logf(fmaxf(acc, 0.f) + 1e-6f);
    lg[idx] = (rl + gum[(size_t)bh * NB * NB + idx]) * (1.f / 0.75f);
  }
  __syncthreads();

  if (d < NB) {
    int u = d;
    float m = -3.0e38f;
    for (int j = 0; j < NB; ++j) m = fmaxf(m, lg[u * NB + j]);
    float s = 0.f;
    for (int j = 0; j < NB; ++j) s += __expf(lg[u * NB + j] - m);
    float inv = 1.f / s;
    for (int j = 0; j < NB; ++j) {
      float val = (j < u) ? __expf(lg[u * NB + j] - m) * inv : 0.f;  // tril(k=-1)
      R[(size_t)bh * NB * NB + u * NB + j] = val;
    }
  }
}

// ---------------------------------------------------------------------------
// Kernel 2: fused reorder + bucketed attention. One block per (bucket, seq).
// ---------------------------------------------------------------------------
__global__ __launch_bounds__(256) void attn_kernel(
    const float* __restrict__ q, const float* __restrict__ k,
    const float* __restrict__ v, const float* __restrict__ R,
    float* __restrict__ out) {
  extern __shared__ __align__(128) char smem[];
  char*  smQ = smem + SM_Q;
  char*  smK = smem + SM_K;
  char*  smP = smem + SM_K;   // overlays K' (dead after S phase)
  char*  smV = smem + SM_V;
  float* smS = (float*)(smem + SM_S);
  float* rr  = (float*)(smem + SM_R);

  const int u   = blockIdx.x;               // bucket
  const int bh  = blockIdx.y;               // b*H + h
  const int h   = bh & (NH - 1);
  const int rot = (h >= HH) ? ROT : 0;
  const int tid = threadIdx.x;
  const int lane = tid & 31;
  const int wv   = tid >> 5;                // wave 0..7, owns rows 16*wv..16*wv+15
  const size_t base = (size_t)bh * T_LEN;

  if (tid < NB) rr[tid] = R[(size_t)bh * NB * NB + u * NB + tid];
  __syncthreads();

  // ---- stage Q as bf16 A-fragments (tiles (mt, kc), 1KB each) ----
  for (int it = 0; it < 8; ++it) {
    int qt = tid + it * 256;                // 0..2047
    int m  = qt >> 4;                       // row 0..127
    int d0 = (qt & 15) << 3;                // 8-wide d group
    int tp = (u * BS + m + rot) & (T_LEN - 1);
    const float* qp = q + (base + tp) * DH + d0;
    unsigned int p0 = f2bf2(qp[0], qp[1]), p1 = f2bf2(qp[2], qp[3]);
    unsigned int p2 = f2bf2(qp[4], qp[5]), p3 = f2bf2(qp[6], qp[7]);
    int gl = (d0 & 31) >> 3;
    int lanep = (m & 15) + ((gl & 2) << 3);
    int eb = (gl & 1) ? 16 : 0;
    *(uint4*)(smQ + (((m >> 4) << 2) + (d0 >> 5)) * 1024 + lanep * 32 + eb) =
        make_uint4(p0, p1, p2, p3);
  }

  // ---- stage K' = [R-reordered bucket | raw bucket] as B-fragments ----
  for (int it = 0; it < 8; ++it) {
    int kt = tid + it * 256;                // 0..2047
    int j  = kt >> 3;                       // K' row 0..255
    int d0 = (kt & 7) << 4;                 // 16-wide d group
    float vals[16];
    if (j >= BS) {                          // raw half
      int tp = (u * BS + (j - BS) + rot) & (T_LEN - 1);
      const float* kp = k + (base + tp) * DH + d0;
#pragma unroll
      for (int i = 0; i < 16; ++i) vals[i] = kp[i];
    } else {                                // reordered half: sum_v R[u,v]*K[v]
#pragma unroll
      for (int i = 0; i < 16; ++i) vals[i] = 0.f;
      for (int vv = 0; vv < u; ++vv) {
        float r = rr[vv];
        int tp = (vv * BS + j + rot) & (T_LEN - 1);
        const float* kp = k + (base + tp) * DH + d0;
#pragma unroll
        for (int i = 0; i < 16; ++i) vals[i] += r * kp[i];
      }
    }
    int lanep = (j & 15) + ((d0 & 31) ? 16 : 0);
    char* dst = smK + (((j >> 4) << 2) + (d0 >> 5)) * 1024 + lanep * 32;
    *(uint4*)dst = make_uint4(f2bf2(vals[0], vals[1]),  f2bf2(vals[2], vals[3]),
                              f2bf2(vals[4], vals[5]),  f2bf2(vals[6], vals[7]));
    *(uint4*)(dst + 16) = make_uint4(f2bf2(vals[8], vals[9]),  f2bf2(vals[10], vals[11]),
                                     f2bf2(vals[12], vals[13]), f2bf2(vals[14], vals[15]));
  }

  // ---- stage V' similarly, as B-fragments for P·V' (k-dim = j) ----
  for (int it = 0; it < 8; ++it) {
    int vt = tid + it * 256;
    int d  = vt >> 4;                       // 0..127
    int j0 = (vt & 15) << 4;                // 16-wide j group
    float vals[16];
    if (j0 >= BS) {
#pragma unroll
      for (int i = 0; i < 16; ++i) {
        int tp = (u * BS + (j0 - BS + i) + rot) & (T_LEN - 1);
        vals[i] = v[(base + tp) * DH + d];
      }
    } else {
#pragma unroll
      for (int i = 0; i < 16; ++i) vals[i] = 0.f;
      for (int vv = 0; vv < u; ++vv) {
        float r = rr[vv];
        int tb = vv * BS + j0 + rot;
#pragma unroll
        for (int i = 0; i < 16; ++i)
          vals[i] += r * v[(base + (size_t)((tb + i) & (T_LEN - 1))) * DH + d];
      }
    }
    int lanep = (d & 15) + ((j0 & 31) ? 16 : 0);
    char* dst = smV + (((d >> 4) << 3) + (j0 >> 5)) * 1024 + lanep * 32;
    *(uint4*)dst = make_uint4(f2bf2(vals[0], vals[1]),  f2bf2(vals[2], vals[3]),
                              f2bf2(vals[4], vals[5]),  f2bf2(vals[6], vals[7]));
    *(uint4*)(dst + 16) = make_uint4(f2bf2(vals[8], vals[9]),  f2bf2(vals[10], vals[11]),
                                     f2bf2(vals[12], vals[13]), f2bf2(vals[14], vals[15]));
  }
  __syncthreads();

  // ---- S = Q · K'^T : wave wv computes its 16-row strip, stages f32 in LDS ----
  v16bf qa[4];
#pragma unroll
  for (int kc = 0; kc < 4; ++kc)
    qa[kc] = ld_frag(smQ + (wv * 4 + kc) * 1024 + lane * 32);

  for (int jt = 0; jt < 16; ++jt) {
    v8f acc = {};
#pragma unroll
    for (int kc = 0; kc < 4; ++kc) {
      v16bf b = ld_frag(smK + (jt * 4 + kc) * 1024 + lane * 32);
      acc = __builtin_amdgcn_wmma_f32_16x16x32_bf16(
          false, qa[kc], false, b, (short)0, acc, false, false);
    }
    // C/D layout: lanes 0-15 -> row p, lanes 16-31 -> row p+8; col = lane&15
    int colb = jt * 16 + (lane & 15);
    int rowb = wv * 16 + ((lane >> 4) << 3);
    float* sp = smS + rowb * 256 + colb;
#pragma unroll
    for (int p = 0; p < 8; ++p) sp[p * 256] = acc[p];
  }
  __syncthreads();   // all waves done with smK before P overlays it

  // ---- masked softmax over 256 cols; write P as bf16 A-fragments ----
  const bool special = (h >= HH) && (u == NB - 1);
  for (int lr = 0; lr < 16; ++lr) {
    int i  = wv * 16 + lr;                  // row within bucket
    int j0 = lane * 8;                      // 8 cols per lane
    const float* sp = smS + i * 256 + j0;
    float s8[8]; bool vis[8];
    float mx = -3.0e38f;
#pragma unroll
    for (int p = 0; p < 8; ++p) {
      int j = j0 + p;
      bool viz;
      if (special) viz = (i == 0) ? (j <= BS) : (j >= BS + 1 && j <= BS + i);
      else         viz = (j < BS) || ((j - BS) <= i);
      float val = sp[p] * SCALE;
      s8[p] = val; vis[p] = viz;
      if (viz) mx = fmaxf(mx, val);
    }
#pragma unroll
    for (int off = 16; off; off >>= 1) mx = fmaxf(mx, __shfl_xor(mx, off, 32));
    float sum = 0.f; float pv[8];
#pragma unroll
    for (int p = 0; p < 8; ++p) {
      float e = vis[p] ? __expf(s8[p] - mx) : 0.f;
      pv[p] = e; sum += e;
    }
#pragma unroll
    for (int off = 16; off; off >>= 1) sum += __shfl_xor(sum, off, 32);
    float inv = 1.f / sum;
    int kc = j0 >> 5;
    int gl = (j0 & 31) >> 3;
    int lanep = lr + ((gl & 2) << 3);
    int eb = (gl & 1) ? 16 : 0;
    *(uint4*)(smP + (wv * 8 + kc) * 1024 + lanep * 32 + eb) =
        make_uint4(f2bf2(pv[0] * inv, pv[1] * inv), f2bf2(pv[2] * inv, pv[3] * inv),
                   f2bf2(pv[4] * inv, pv[5] * inv), f2bf2(pv[6] * inv, pv[7] * inv));
  }

  // ---- O = P · V' ; store with inverse rotation ----
  v16bf pa[8];
#pragma unroll
  for (int kc = 0; kc < 8; ++kc)
    pa[kc] = ld_frag(smP + (wv * 8 + kc) * 1024 + lane * 32);

  for (int nt = 0; nt < 8; ++nt) {
    v8f acc = {};
#pragma unroll
    for (int kc = 0; kc < 8; ++kc) {
      v16bf b = ld_frag(smV + (nt * 8 + kc) * 1024 + lane * 32);
      acc = __builtin_amdgcn_wmma_f32_16x16x32_bf16(
          false, pa[kc], false, b, (short)0, acc, false, false);
    }
    int colb = nt * 16 + (lane & 15);
    int rowb = wv * 16 + ((lane >> 4) << 3);
#pragma unroll
    for (int p = 0; p < 8; ++p) {
      int i  = rowb + p;
      int tp = (u * BS + i + rot) & (T_LEN - 1);   // rotate_right on output
      out[(base + tp) * DH + colb] = acc[p];
    }
  }
}

// ---------------------------------------------------------------------------
extern "C" void kernel_launch(void* const* d_in, const int* in_sizes, int n_in,
                              void* d_out, int out_size, void* d_ws, size_t ws_size,
                              hipStream_t stream) {
  (void)in_sizes; (void)n_in; (void)out_size; (void)ws_size;
  const float* q     = (const float*)d_in[0];
  const float* k     = (const float*)d_in[1];
  const float* v     = (const float*)d_in[2];
  const float* sortw = (const float*)d_in[3];
  const float* gum   = (const float*)d_in[4];
  float* out = (float*)d_out;
  float* R   = (float*)d_ws;                       // 32*32*32 f32 = 128 KB

  sort_r_kernel<<<32, 128, 0, stream>>>(k, sortw, gum, R);
  attn_kernel<<<dim3(NB, 32), 256, SM_TOTAL, stream>>>(q, k, v, R, out);
}
